// AttentionFuserV3_65060164600007
// MI455X (gfx1250) — matrix operand order repaired and checked by reading, not compiled, not run on hardware
//
#include <hip/hip_runtime.h>
#include <hip/hip_bf16.h>

// ---------------------------------------------------------------------------
// AttentionFuserV3 for MI455X (gfx1250, wave32, WMMA).
// GEMMs: v_wmma_f32_16x16x32_bf16, double-buffered LDS, async global->LDS.
// ---------------------------------------------------------------------------

#define USE_ASYNC 1            // set 0 to fall back to sync global->LDS copies

typedef __attribute__((ext_vector_type(16))) __bf16 bf16x16;
typedef __attribute__((ext_vector_type(4)))  __bf16 bf16x4;
typedef __attribute__((ext_vector_type(8)))  float  f32x8;

union Frag16 { bf16x16 v; uint4 q[2]; };

// LDS tile: 128 rows x 32 bf16, padded to 40 bf16 (80B) per row.
// 80B row stride => 16B-aligned ds_load_b128 and conflict-free lane->bank map.
#define LDS_STRIDE 40
#define LDS_TILE   (128 * LDS_STRIDE)

// 16B async copy: global -> LDS, tracked by ASYNCcnt (CDNA5 path).
__device__ __forceinline__ void cp16_g2l(const __bf16* gsrc, __bf16* ldst)
{
#if USE_ASYNC
    asm volatile("global_load_async_to_lds_b128 %0, %1, off"
                 :: "v"((__attribute__((address_space(3))) __bf16*)ldst),
                    "v"((const __attribute__((address_space(1))) __bf16*)gsrc)
                 : "memory");
#else
    *(uint4*)ldst = *(const uint4*)gsrc;
#endif
}

__device__ __forceinline__ void wait_async_copies()
{
#if USE_ASYNC
#if defined(__has_builtin) && __has_builtin(__builtin_amdgcn_s_wait_asynccnt)
    __builtin_amdgcn_s_wait_asynccnt(0);
#else
    asm volatile("s_wait_asynccnt 0" ::: "memory");
#endif
#endif
}

// ---------------------------------------------------------------------------
// Batched GEMM: C[b] (MxN) = A[b] (MxK, row-major bf16) * op(B[b])
//   TRANSB==true : B memory is [N][K] row-major (C = A*B^T) -- weights, Q@K^T
//   TRANSB==false: B memory is [K][N] row-major (C = A*B)   -- attn @ V
// Block tile 128x128, Ktile 32, 256 threads = 8 waves (4 M x 2 N),
// each wave: 2x4 grid of 16x16 WMMA tiles. Double-buffered LDS,
// one barrier per k-tile, async copies overlapped with WMMA.
// ---------------------------------------------------------------------------
template<bool BF16_OUT, bool TRANSB>
__global__ __launch_bounds__(256)
void gemm_wmma_bf16(const __bf16* __restrict__ A, const __bf16* __restrict__ B,
                    float* __restrict__ Cf, __bf16* __restrict__ Cb,
                    int M, int Nn, int K, int lda, int ldb, int ldc,
                    long long sA, long long sB, long long sC)
{
    __shared__ __bf16 ldsA[2][LDS_TILE];
    __shared__ __bf16 ldsB[2][LDS_TILE];

    const int bz = blockIdx.z;
    A += (long long)bz * sA;
    B += (long long)bz * sB;
    const int rowBase = blockIdx.y * 128;
    const int colBase = blockIdx.x * 128;

    const int tid   = threadIdx.x;
    const int lane  = tid & 31;
    const int w     = tid >> 5;        // 0..7
    const int waveM = w >> 1;          // 0..3 -> 32-row patch
    const int waveN = w & 1;           // 0..1 -> 64-col patch
    const int lrow  = lane & 15;       // row/col inside a 16x16 tile
    const int kh    = lane >> 4;       // K-half select

    // per-thread staging coordinates (512 x 16B tasks over 256 threads)
    int rS[2], gS[2];                  // row-major 16B-chunk tasks (A, B^T)
    int kT[2], gT[2];                  // transpose tasks (32 k x 16 n-groups)
#pragma unroll
    for (int i = 0; i < 2; ++i) {
        int task = tid + i * 256;
        rS[i] = task >> 2;  gS[i] = (task & 3) << 3;
        kT[i] = task >> 4;  gT[i] = (task & 15) << 3;
    }

    // --- staging helpers ---
    auto stageA = [&](int kt, int buf) {
#pragma unroll
        for (int i = 0; i < 2; ++i)
            cp16_g2l(&A[(long long)(rowBase + rS[i]) * lda + kt + gS[i]],
                     &ldsA[buf][rS[i] * LDS_STRIDE + gS[i]]);
    };
    auto stageBt = [&](int kt, int buf) {   // TRANSB: B memory [N][K]
#pragma unroll
        for (int i = 0; i < 2; ++i)
            cp16_g2l(&B[(long long)(colBase + rS[i]) * ldb + kt + gS[i]],
                     &ldsB[buf][rS[i] * LDS_STRIDE + gS[i]]);
    };
    auto loadBn = [&](int kt, uint4* regs) {  // !TRANSB: B memory [K][N]
#pragma unroll
        for (int i = 0; i < 2; ++i)
            regs[i] = *(const uint4*)(&B[(long long)(kt + kT[i]) * ldb + colBase + gT[i]]);
    };
    auto scatterBn = [&](const uint4* regs, int buf) {  // transpose into LDS
#pragma unroll
        for (int i = 0; i < 2; ++i) {
            const __bf16* pv = (const __bf16*)&regs[i];
#pragma unroll
            for (int j = 0; j < 8; ++j)
                ldsB[buf][(gT[i] + j) * LDS_STRIDE + kT[i]] = pv[j];
        }
    };

    f32x8 acc[2][4];
#pragma unroll
    for (int mi = 0; mi < 2; ++mi)
#pragma unroll
        for (int ni = 0; ni < 4; ++ni)
#pragma unroll
            for (int r = 0; r < 8; ++r) acc[mi][ni][r] = 0.f;

    // ---- prologue: fill buffer 0 ----
    stageA(0, 0);
    if (TRANSB) {
        stageBt(0, 0);
    } else {
        uint4 br[2];
        loadBn(0, br);
        scatterBn(br, 0);
    }
    wait_async_copies();
    __syncthreads();

    const int nk = K >> 5;
    for (int t = 0; t < nk; ++t) {
        const int cur = t & 1, nxt = cur ^ 1;
        const bool more = (t + 1) < nk;
        uint4 bregs[2];

        // issue next tile's copies before compute (latency overlap)
        if (more) {
            stageA((t + 1) << 5, nxt);
            if (TRANSB) stageBt((t + 1) << 5, nxt);
            else        loadBn((t + 1) << 5, bregs);
        }

        // ---- fragments per ISA layout ----
        const __bf16* lA = ldsA[cur];
        const __bf16* lB = ldsB[cur];
        Frag16 fa[2], fb[4];
#pragma unroll
        for (int mi = 0; mi < 2; ++mi) {
            int r = waveM * 32 + mi * 16 + lrow;
            // elements 0..7 : K = 8*kh.. ; elements 8..15 : K = 16+8*kh..
            fa[mi].q[0] = *(const uint4*)(&lA[r * LDS_STRIDE + 8 * kh]);
            fa[mi].q[1] = *(const uint4*)(&lA[r * LDS_STRIDE + 16 + 8 * kh]);
        }
#pragma unroll
        for (int ni = 0; ni < 4; ++ni) {
            int c = waveN * 64 + ni * 16 + lrow;
            // elements 0..7 : K = 8*kh.. ; elements 8..15 : K = 16+8*kh..
            fb[ni].q[0] = *(const uint4*)(&lB[c * LDS_STRIDE + 8 * kh]);
            fb[ni].q[1] = *(const uint4*)(&lB[c * LDS_STRIDE + 16 + 8 * kh]);
        }

#pragma unroll
        for (int mi = 0; mi < 2; ++mi)
#pragma unroll
            for (int ni = 0; ni < 4; ++ni)
                acc[mi][ni] = __builtin_amdgcn_wmma_f32_16x16x32_bf16(
                    false, fa[mi].v, false, fb[ni].v,
                    (short)0, acc[mi][ni], false, false);

        if (more && !TRANSB) scatterBn(bregs, nxt);
        if (more) wait_async_copies();
        __syncthreads();
    }

    // ---- epilogue: C/D layout -> VGPR r holds M = r + 8*kh, N = lrow ----
    const long long cbase = (long long)bz * sC;
#pragma unroll
    for (int mi = 0; mi < 2; ++mi)
#pragma unroll
        for (int ni = 0; ni < 4; ++ni)
#pragma unroll
            for (int r = 0; r < 8; ++r) {
                int row = rowBase + waveM * 32 + mi * 16 + 8 * kh + r;
                int col = colBase + waveN * 64 + ni * 16 + lrow;
                float v = acc[mi][ni][r];
                if (BF16_OUT) Cb[cbase + (long long)row * ldc + col] = (__bf16)v;
                else          Cf[cbase + (long long)row * ldc + col] = v;
            }
}

// ---------------------------------------------------------------------------
// Row softmax: fp32 scores row [Lrow<=1024] -> bf16 attention row.
// One block per row; row cached in registers (single global read pass).
// ---------------------------------------------------------------------------
__global__ __launch_bounds__(256)
void softmax_rows(const float* __restrict__ S, __bf16* __restrict__ Aout, int Lrow)
{
    __shared__ float red[256];
    const long long row = blockIdx.x;
    const float* s = S + row * Lrow;
    __bf16* o = Aout + row * Lrow;
    const int tid = threadIdx.x;

    float v[4];
    float m = -3.402823466e38f;
#pragma unroll
    for (int i = 0; i < 4; ++i) {
        int idx = tid + i * 256;
        v[i] = (idx < Lrow) ? s[idx] : -3.402823466e38f;
        m = fmaxf(m, v[i]);
    }
    red[tid] = m; __syncthreads();
    for (int st = 128; st > 0; st >>= 1) {
        if (tid < st) red[tid] = fmaxf(red[tid], red[tid + st]);
        __syncthreads();
    }
    m = red[0]; __syncthreads();

    float sum = 0.f;
#pragma unroll
    for (int i = 0; i < 4; ++i) { v[i] = expf(v[i] - m); sum += v[i]; }
    red[tid] = sum; __syncthreads();
    for (int st = 128; st > 0; st >>= 1) {
        if (tid < st) red[tid] += red[tid + st];
        __syncthreads();
    }
    const float inv = 1.f / red[0];
#pragma unroll
    for (int i = 0; i < 4; ++i) {
        int idx = tid + i * 256;
        if (idx < Lrow) o[idx] = (__bf16)(v[i] * inv);
    }
}

// ---------------------------------------------------------------------------
// hidden = L2normalize(tanh(H)) over D (<=512) ; pack c2 = [hidden | x] bf16.
// One block per (n,l) row; tanh values cached in registers.
// ---------------------------------------------------------------------------
__global__ __launch_bounds__(256)
void tanh_norm_pack(const float* __restrict__ H, const float* __restrict__ X,
                    __bf16* __restrict__ C2, int Dd)
{
    __shared__ float red[256];
    const long long row = blockIdx.x;
    const float* h = H + row * Dd;
    const float* x = X + row * Dd;
    __bf16* c = C2 + row * (2 * (long long)Dd);
    const int tid = threadIdx.x;

    float t[2];
    float ss = 0.f;
#pragma unroll
    for (int i = 0; i < 2; ++i) {
        int idx = tid + i * 256;
        t[i] = (idx < Dd) ? tanhf(h[idx]) : 0.f;
        ss += t[i] * t[i];
    }
    red[tid] = ss; __syncthreads();
    for (int st = 128; st > 0; st >>= 1) {
        if (tid < st) red[tid] += red[tid + st];
        __syncthreads();
    }
    const float inv = 1.f / fmaxf(sqrtf(red[0]), 1e-12f);
#pragma unroll
    for (int i = 0; i < 2; ++i) {
        int idx = tid + i * 256;
        if (idx < Dd) {
            c[idx]      = (__bf16)(t[i] * inv);
            c[Dd + idx] = (__bf16)x[idx];
        }
    }
}

// ---------------------------------------------------------------------------
// out[n,d] = mean over L of E[n,l,d]
// ---------------------------------------------------------------------------
__global__ __launch_bounds__(256)
void mean_over_L(const float* __restrict__ E, float* __restrict__ O,
                 int Lrow, int Dd, int total)
{
    int idx = blockIdx.x * blockDim.x + threadIdx.x;
    if (idx >= total) return;
    int n = idx / Dd, d = idx % Dd;
    const float* e = E + (long long)n * Lrow * Dd + d;
    float s = 0.f;
    for (int l = 0; l < Lrow; ++l) s += e[(long long)l * Dd];
    O[idx] = s / (float)Lrow;
}

// fp32 -> bf16, 4 elements per thread (all our sizes are multiples of 4)
__global__ __launch_bounds__(256)
void f32_to_bf16_v4(const float* __restrict__ in, __bf16* __restrict__ out,
                    long long n4)
{
    long long i = (long long)blockIdx.x * 256 + threadIdx.x;
    if (i < n4) {
        float4 f = ((const float4*)in)[i];
        bf16x4 b = { (__bf16)f.x, (__bf16)f.y, (__bf16)f.z, (__bf16)f.w };
        ((bf16x4*)out)[i] = b;
    }
}

// ---------------------------------------------------------------------------
// Host-side orchestration
// ---------------------------------------------------------------------------
static inline void launch_gemm(const __bf16* A, const __bf16* B,
                               float* Cf, __bf16* Cb,
                               int M, int Nn, int K, int lda, int ldb, int ldc,
                               long long sA, long long sB, long long sC,
                               int nb, int transB, hipStream_t stream)
{
    dim3 grid(Nn / 128, M / 128, nb), block(256);
    if (Cb) {
        if (transB)
            gemm_wmma_bf16<true, true><<<grid, block, 0, stream>>>(
                A, B, nullptr, Cb, M, Nn, K, lda, ldb, ldc, sA, sB, sC);
        else
            gemm_wmma_bf16<true, false><<<grid, block, 0, stream>>>(
                A, B, nullptr, Cb, M, Nn, K, lda, ldb, ldc, sA, sB, sC);
    } else {
        if (transB)
            gemm_wmma_bf16<false, true><<<grid, block, 0, stream>>>(
                A, B, Cf, nullptr, M, Nn, K, lda, ldb, ldc, sA, sB, sC);
        else
            gemm_wmma_bf16<false, false><<<grid, block, 0, stream>>>(
                A, B, Cf, nullptr, M, Nn, K, lda, ldb, ldc, sA, sB, sC);
    }
}

extern "C" void kernel_launch(void* const* d_in, const int* in_sizes, int n_in,
                              void* d_out, int out_size, void* d_ws, size_t ws_size,
                              hipStream_t stream)
{
    const int NB = 32, L = 1024, D = 512;
    const long long LD  = (long long)L * D;
    const long long LL  = (long long)L * L;
    const long long L2D = (long long)L * 2 * D;
    const long long L4D = (long long)L * 4 * D;

    const float* x   = (const float*)d_in[0];      // [N,L,D]
    const float* W1  = (const float*)d_in[1];      // [D,D]
    const float* Wo1 = (const float*)d_in[2];      // [D,2D]
    const float* W2  = (const float*)d_in[3];      // [2D,2D]
    const float* Wo2 = (const float*)d_in[4];      // [D,4D]
    float* out = (float*)d_out;                    // [N,D,1,1]

    // ---- workspace layout ----
    char* ws = (char*)d_ws;
    size_t off = 0;
    auto alloc = [&](size_t bytes) -> char* {
        char* p = ws + off;
        off += (bytes + 255) & ~(size_t)255;
        return p;
    };
    __bf16* xb    = (__bf16*)alloc((size_t)NB * LD * 2);
    __bf16* w1b   = (__bf16*)alloc((size_t)D * D * 2);
    __bf16* wo1b  = (__bf16*)alloc((size_t)D * 2 * D * 2);
    __bf16* w2b   = (__bf16*)alloc((size_t)2 * D * 2 * D * 2);
    __bf16* wo2b  = (__bf16*)alloc((size_t)D * 4 * D * 2);
    float*  score = (float*) alloc((size_t)NB * LL * 4);   // reused by both layers
    __bf16* attnb = (__bf16*)alloc((size_t)NB * LL * 2);   // reused
    __bf16* comb1 = (__bf16*)alloc((size_t)NB * L2D * 2);  // [mix | q1]
    float*  hid   = (float*) alloc((size_t)NB * LD * 4);   // preact / emb (reused)
    __bf16* c2    = (__bf16*)alloc((size_t)NB * L2D * 2);  // [hidden | x]
    __bf16* comb2 = (__bf16*)alloc((size_t)NB * L4D * 2);  // [mix2 | q2]
    (void)ws_size; (void)in_sizes; (void)n_in; (void)out_size;

    // ---- fp32 -> bf16 conversions (vectorized x4) ----
    auto cvt = [&](const float* src, __bf16* dst, long long n) {
        long long n4 = n >> 2;
        f32_to_bf16_v4<<<(unsigned)((n4 + 255) / 256), 256, 0, stream>>>(src, dst, n4);
    };
    cvt(x,   xb,   (long long)NB * LD);
    cvt(W1,  w1b,  (long long)D * D);
    cvt(Wo1, wo1b, (long long)D * 2 * D);
    cvt(W2,  w2b,  (long long)2 * D * 2 * D);
    cvt(Wo2, wo2b, (long long)D * 4 * D);

    // ================= stage 1 =================
    // q1 = x @ W1^T  -> comb1 right half (bf16)
    launch_gemm(xb, w1b, nullptr, comb1 + D,
                L, D, D, D, D, 2 * D, LD, 0, L2D, NB, 1, stream);
    // scores = q1 @ x^T (fp32)
    launch_gemm(comb1 + D, xb, score, nullptr,
                L, L, D, 2 * D, D, L, L2D, LD, LL, NB, 1, stream);
    softmax_rows<<<NB * L, 256, 0, stream>>>(score, attnb, L);
    // mix = attn @ x -> comb1 left half
    launch_gemm(attnb, xb, nullptr, comb1,
                L, D, L, L, D, 2 * D, LL, LD, L2D, NB, 0, stream);
    // hidden_pre = [mix|q1] @ Wo1^T (fp32)
    launch_gemm(comb1, wo1b, hid, nullptr,
                L, D, 2 * D, 2 * D, 2 * D, D, L2D, 0, LD, NB, 1, stream);
    // c2 = [ L2norm(tanh(hidden_pre)) | x ] (bf16)
    tanh_norm_pack<<<NB * L, 256, 0, stream>>>(hid, x, c2, D);

    // ================= stage 2 =================
    // q2 = c2 @ W2^T -> comb2 right half
    launch_gemm(c2, w2b, nullptr, comb2 + 2 * D,
                L, 2 * D, 2 * D, 2 * D, 2 * D, 4 * D, L2D, 0, L4D, NB, 1, stream);
    // scores2 = q2 @ c2^T (fp32)
    launch_gemm(comb2 + 2 * D, c2, score, nullptr,
                L, L, 2 * D, 4 * D, 2 * D, L, L4D, L2D, LL, NB, 1, stream);
    softmax_rows<<<NB * L, 256, 0, stream>>>(score, attnb, L);
    // mix2 = attn2 @ c2 -> comb2 left half
    launch_gemm(attnb, c2, nullptr, comb2,
                L, 2 * D, L, L, 2 * D, 4 * D, LL, L2D, L4D, NB, 0, stream);
    // emb = [mix2|q2] @ Wo2^T (fp32, into hid buffer)
    launch_gemm(comb2, wo2b, hid, nullptr,
                L, D, 4 * D, 4 * D, 4 * D, D, L4D, 0, LD, NB, 1, stream);

    // mean over L -> out [N*D]
    const int total = NB * D;
    mean_over_L<<<(total + 255) / 256, 256, 0, stream>>>(hid, out, L, D, total);
}
